// ChiralRnnBase_34179349741810
// MI455X (gfx1250) — matrix operand order repaired and checked by reading, not compiled, not run on hardware
//
#include <hip/hip_runtime.h>
#include <math.h>

#define T_STEPS 512
#define BATCH   64
#define DIN     128
#define HDIM    1024

typedef __attribute__((ext_vector_type(16))) __bf16 v16bf;
typedef __attribute__((ext_vector_type(8)))  __bf16 v8bf;
typedef __attribute__((ext_vector_type(8)))  float  v8f;

__device__ __forceinline__ float sigf(float x) {
    return 1.0f / (1.0f + __expf(-x));
}

// ---------------- small utility kernels ----------------

__global__ void cvt_f32_bf16(const float* __restrict__ src, __bf16* __restrict__ dst, int n) {
    int i = blockIdx.x * blockDim.x + threadIdx.x;
    if (i < n) dst[i] = (__bf16)src[i];
}

__global__ void zero_u32(unsigned* __restrict__ p, int n) {
    int i = blockIdx.x * blockDim.x + threadIdx.x;
    if (i < n) p[i] = 0u;
}

__global__ void bias_comb(const float* __restrict__ a, const float* __restrict__ b,
                          float* __restrict__ o, int n) {
    int i = blockIdx.x * blockDim.x + threadIdx.x;
    if (i < n) o[i] = a[i] + b[i];
}

__global__ void copy_f32(const float* __restrict__ s, float* __restrict__ d, int n) {
    int i = blockIdx.x * blockDim.x + threadIdx.x;
    if (i < n) d[i] = s[i];
}

// ---------------- fused LSTM step ----------------
// One timestep, one layer:  gates = x_t @ w_ih^T + h_{t-1} @ w_hh^T + bias,
// then the elementwise cell update -- all fused.
//
// Grid: HDIM/16 = 64 blocks, 512 threads = 16 waves.
//   wave = 4*ks + m :  m  = batch tile (16 rows of B=64)
//                      ks = K-split (4-way split of the K=KIN+HDIM reduction)
// Each wave accumulates 4 WMMA C-tiles (gates i,f,g,o) over its K quarter,
// partial sums are reduced through LDS, and the ks==0 waves apply the
// nonlinearities + c/h update directly on the reduced accumulators.
// K loop is software-pipelined with a 2-deep register fragment buffer.
template <int KIN>
__global__ __launch_bounds__(512)
void lstm_step(const __bf16* __restrict__ xin,    // [BATCH, KIN]   bf16 input at t
               const __bf16* __restrict__ hprev,  // [BATCH, HDIM]  bf16 h_{t-1}
               const __bf16* __restrict__ wih,    // [4*HDIM, KIN]  bf16
               const __bf16* __restrict__ whh,    // [4*HDIM, HDIM] bf16
               const float*  __restrict__ bias,   // [4*HDIM] (b_ih + b_hh)
               float*        __restrict__ c,      // [BATCH, HDIM]  f32, in/out
               __bf16*       __restrict__ hnext,  // [BATCH, HDIM]  bf16 out
               float*        __restrict__ h32,    // [BATCH, HDIM]  f32 out
               float*        __restrict__ yout)   // optional f32 out (layer-1 -> d_out)
{
    constexpr int NCH = (KIN + HDIM) / 32;  // total K chunks of 32
    constexpr int NIT = NCH / 4;            // chunks per wave (9 or 16, uniform)

    const int lane = threadIdx.x & 31;
    const int wave = threadIdx.x >> 5;      // 0..15
    const int m    = wave & 3;              // batch tile 0..3
    const int ks   = wave >> 2;             // K-split 0..3
    const int j0   = blockIdx.x * 16;       // 16-column tile in H
    const int row  = lane & 15;
    const bool hi  = lane >= 16;
    const int koffA = hi ? 16 : 0;          // A-fragment K offset (contiguous k = 16h + s)
    const int koffB = hi ? 16 : 0;          // B-fragment K offset (contiguous k = 16h + s, as A)

    // per-wave base pointers
    const __bf16* Ax = xin   + (size_t)(m * 16 + row) * KIN;
    const __bf16* Ah = hprev + (size_t)(m * 16 + row) * HDIM;
    const __bf16* Bx[4];
    const __bf16* Bh[4];
#pragma unroll
    for (int g = 0; g < 4; ++g) {
        Bx[g] = wih + (size_t)(g * HDIM + j0 + row) * KIN;
        Bh[g] = whh + (size_t)(g * HDIM + j0 + row) * HDIM;
    }

    struct Frag { v16bf a; v16bf b[4]; };
    auto loadFrag = [&](int chunk, Frag& f) {
        const int k = chunk * 32;
        const bool inx = (k < KIN);
        const int kk = inx ? k : (k - KIN);
        const __bf16* A = (inx ? Ax : Ah) + kk + koffA;
        union { v16bf v; v8bf h[2]; } af;
        af.h[0] = *(const v8bf*)(A);
        af.h[1] = *(const v8bf*)(A + 8);
        f.a = af.v;
#pragma unroll
        for (int g = 0; g < 4; ++g)
            f.b[g] = *(const v16bf*)((inx ? Bx[g] : Bh[g]) + kk + koffB);
    };

    v8f acc[4] = {v8f{}, v8f{}, v8f{}, v8f{}};  // i, f, g, o partials
    Frag fr[2];

    loadFrag(ks, fr[0]);
#pragma unroll
    for (int it = 0; it < NIT; ++it) {
        if (it + 1 < NIT) loadFrag(ks + (it + 1) * 4, fr[(it + 1) & 1]);
        Frag& f = fr[it & 1];
#pragma unroll
        for (int g = 0; g < 4; ++g)
            acc[g] = __builtin_amdgcn_wmma_f32_16x16x32_bf16(
                false, f.a, false, f.b[g], (short)0, acc[g], false, false);
    }

    // ---- cross-K-split reduction through LDS ----
    __shared__ float red[16 * 4 * 256];  // [wave][gate][256] = 64 KB
    {
        float* base = &red[(wave * 4) * 256 + lane * 8];
#pragma unroll
        for (int g = 0; g < 4; ++g)
            *(v8f*)(base + g * 256) = acc[g];
    }
    __syncthreads();

    if (ks == 0) {
#pragma unroll
        for (int s = 1; s < 4; ++s) {
            const float* base = &red[((s * 4 + m) * 4) * 256 + lane * 8];
#pragma unroll
            for (int g = 0; g < 4; ++g)
                acc[g] += *(const v8f*)(base + g * 256);
        }

        // ---- elementwise LSTM cell update ----
        const int jcol = j0 + (lane & 15);       // C/D layout: N = lane&15
        const float bi  = bias[0 * HDIM + jcol];
        const float bf_ = bias[1 * HDIM + jcol];
        const float bg  = bias[2 * HDIM + jcol];
        const float bo  = bias[3 * HDIM + jcol];

#pragma unroll
        for (int e = 0; e < 8; ++e) {
            const int b   = m * 16 + e + (hi ? 8 : 0);  // C/D layout: M = e (+8 hi lanes)
            const int idx = b * HDIM + jcol;
            float iv = sigf(acc[0][e] + bi);
            float fv = sigf(acc[1][e] + bf_);
            float gv = tanhf(acc[2][e] + bg);
            float ov = sigf(acc[3][e] + bo);
            float cn = fv * c[idx] + iv * gv;
            float hn = ov * tanhf(cn);
            c[idx]     = cn;
            hnext[idx] = (__bf16)hn;
            h32[idx]   = hn;
            if (yout) yout[idx] = hn;
        }
    }
}

// ---------------- host launcher ----------------

extern "C" void kernel_launch(void* const* d_in, const int* in_sizes, int n_in,
                              void* d_out, int out_size, void* d_ws, size_t ws_size,
                              hipStream_t stream) {
    const float* x     = (const float*)d_in[0];
    const float* w_ih0 = (const float*)d_in[1];
    const float* w_hh0 = (const float*)d_in[2];
    const float* b_ih0 = (const float*)d_in[3];
    const float* b_hh0 = (const float*)d_in[4];
    const float* w_ih1 = (const float*)d_in[5];
    const float* w_hh1 = (const float*)d_in[6];
    const float* b_ih1 = (const float*)d_in[7];
    const float* b_hh1 = (const float*)d_in[8];
    float* out = (float*)d_out;  // [T*B*H] out, then [2*B*H] hs, then [2*B*H] cs

    // ---- carve workspace (all 256B aligned) ----
    char* p = (char*)d_ws;
    auto carve = [&](size_t bytes) {
        void* r = (void*)p;
        p += (bytes + 255) & ~(size_t)255;
        return r;
    };
    __bf16* xb      = (__bf16*)carve((size_t)T_STEPS * BATCH * DIN * 2);  // x in bf16
    __bf16* wb_ih0  = (__bf16*)carve((size_t)4 * HDIM * DIN * 2);
    __bf16* wb_hh0  = (__bf16*)carve((size_t)4 * HDIM * HDIM * 2);
    __bf16* wb_ih1  = (__bf16*)carve((size_t)4 * HDIM * HDIM * 2);
    __bf16* wb_hh1  = (__bf16*)carve((size_t)4 * HDIM * HDIM * 2);
    float*  bias0   = (float*)carve((size_t)4 * HDIM * 4);
    float*  bias1   = (float*)carve((size_t)4 * HDIM * 4);
    __bf16* y0b     = (__bf16*)carve((size_t)2 * BATCH * HDIM * 2);  // layer0 h ping-pong (bf16)
    __bf16* hb1     = (__bf16*)carve((size_t)2 * BATCH * HDIM * 2);  // layer1 h ping-pong (bf16)
    float*  c0      = (float*)carve((size_t)BATCH * HDIM * 4);
    float*  c1      = (float*)carve((size_t)BATCH * HDIM * 4);
    float*  h32_0   = (float*)carve((size_t)BATCH * HDIM * 4);
    float*  h32_1   = (float*)carve((size_t)BATCH * HDIM * 4);

    const int BH = BATCH * HDIM;  // 65536
    auto blks = [](int n) { return (n + 255) / 256; };

    // ---- one-time-per-call prep (deterministic) ----
    cvt_f32_bf16<<<blks(T_STEPS * BATCH * DIN), 256, 0, stream>>>(x, xb, T_STEPS * BATCH * DIN);
    cvt_f32_bf16<<<blks(4 * HDIM * DIN), 256, 0, stream>>>(w_ih0, wb_ih0, 4 * HDIM * DIN);
    cvt_f32_bf16<<<blks(4 * HDIM * HDIM), 256, 0, stream>>>(w_hh0, wb_hh0, 4 * HDIM * HDIM);
    cvt_f32_bf16<<<blks(4 * HDIM * HDIM), 256, 0, stream>>>(w_ih1, wb_ih1, 4 * HDIM * HDIM);
    cvt_f32_bf16<<<blks(4 * HDIM * HDIM), 256, 0, stream>>>(w_hh1, wb_hh1, 4 * HDIM * HDIM);
    bias_comb<<<blks(4 * HDIM), 256, 0, stream>>>(b_ih0, b_hh0, bias0, 4 * HDIM);
    bias_comb<<<blks(4 * HDIM), 256, 0, stream>>>(b_ih1, b_hh1, bias1, 4 * HDIM);
    // zero initial h (bf16 ping-pong, 2*BH bf16 = BH dwords each) and c states
    zero_u32<<<blks(BH), 256, 0, stream>>>((unsigned*)y0b, BH);
    zero_u32<<<blks(BH), 256, 0, stream>>>((unsigned*)hb1, BH);
    zero_u32<<<blks(BH), 256, 0, stream>>>((unsigned*)c0, BH);
    zero_u32<<<blks(BH), 256, 0, stream>>>((unsigned*)c1, BH);

    // ---- recurrence: interleave layer0/layer1 per timestep ----
    dim3 sgrid(HDIM / 16), sblk(512);
    for (int t = 0; t < T_STEPS; ++t) {
        const int wr = t & 1, rd = 1 - wr;
        // layer 0: x input = xb[t], h input = y0b[rd], writes y0b[wr]
        lstm_step<DIN><<<sgrid, sblk, 0, stream>>>(
            xb + (size_t)t * BATCH * DIN, y0b + (size_t)rd * BH,
            wb_ih0, wb_hh0, bias0, c0,
            y0b + (size_t)wr * BH, h32_0, (float*)nullptr);
        // layer 1: x input = layer0 output at t (y0b[wr]), h input = hb1[rd]
        lstm_step<HDIM><<<sgrid, sblk, 0, stream>>>(
            y0b + (size_t)wr * BH, hb1 + (size_t)rd * BH,
            wb_ih1, wb_hh1, bias1, c1,
            hb1 + (size_t)wr * BH, h32_1, out + (size_t)t * BH);
    }

    // ---- tail: hs = [h32_0, h32_1], cs = [c0, c1] ----
    float* tail = out + (size_t)T_STEPS * BH;
    copy_f32<<<blks(BH), 256, 0, stream>>>(h32_0, tail + 0 * BH, BH);
    copy_f32<<<blks(BH), 256, 0, stream>>>(h32_1, tail + 1 * BH, BH);
    copy_f32<<<blks(BH), 256, 0, stream>>>(c0, tail + 2 * BH, BH);
    copy_f32<<<blks(BH), 256, 0, stream>>>(c1, tail + 3 * BH, BH);
}
